// SpatialProcessor_11630771438424
// MI455X (gfx1250) — matrix operand order repaired and checked by reading, not compiled, not run on hardware
//
#include <hip/hip_runtime.h>

#define BATCH 8
#define NNODE 2048
#define DIM   128
#define TOPK  20
#define NBR   (TOPK + 1)
#define LN_EPS 1e-3f

typedef __attribute__((ext_vector_type(16))) __bf16 v16bf;
typedef __attribute__((ext_vector_type(8)))  __bf16 v8bf;
typedef __attribute__((ext_vector_type(8)))  float  v8f;

// ---------------- adjacency build ----------------

__global__ void normalize_emb_kernel(const float* __restrict__ emb,
                                     float* __restrict__ nemb) {
  int row = blockIdx.x * blockDim.x + threadIdx.x;
  if (row >= NNODE) return;
  float e[16]; float ss = 0.f;
  #pragma unroll
  for (int c = 0; c < 16; ++c) { e[c] = emb[row * 16 + c]; ss += e[c] * e[c]; }
  float inv = rsqrtf(ss);
  #pragma unroll
  for (int c = 0; c < 16; ++c) nemb[row * 16 + c] = e[c] * inv;
}

__global__ __launch_bounds__(256) void topk_kernel(const float* __restrict__ nemb,
                                                   int* __restrict__ nbr) {
  __shared__ float sims[NNODE];
  __shared__ float rv[256];
  __shared__ int   ri[256];
  __shared__ float erow[16];
  const int i = blockIdx.x;
  const int tid = threadIdx.x;
  if (tid < 16) erow[tid] = nemb[i * 16 + tid];
  __syncthreads();
  for (int j = tid; j < NNODE; j += 256) {
    float d = 0.f;
    #pragma unroll
    for (int c = 0; c < 16; ++c) d += erow[c] * nemb[j * 16 + c];
    sims[j] = (j == i) ? -3.0e38f : d;   // mask diagonal
  }
  __syncthreads();
  for (int t = 0; t < TOPK; ++t) {
    float bv = -3.0e38f; int bi = 0;
    for (int j = tid; j < NNODE; j += 256)
      if (sims[j] > bv) { bv = sims[j]; bi = j; }
    rv[tid] = bv; ri[tid] = bi;
    __syncthreads();
    for (int s = 128; s > 0; s >>= 1) {
      if (tid < s && rv[tid + s] > rv[tid]) { rv[tid] = rv[tid + s]; ri[tid] = ri[tid + s]; }
      __syncthreads();
    }
    if (tid == 0) { nbr[i * TOPK + t] = ri[0]; sims[ri[0]] = -3.0e38f; }
    __syncthreads();
  }
}

// ---------------- precision conversion ----------------

__global__ void f32_to_bf16_kernel(const float* __restrict__ in,
                                   __bf16* __restrict__ out, int n) {
  int idx = blockIdx.x * blockDim.x + threadIdx.x;
  if (idx < n) out[idx] = (__bf16)in[idx];
}

// store weight transposed: wt[n*DIM + k] = W[k*DIM + n]  (B-frag wants contiguous K per column)
__global__ void weight_to_bf16T_kernel(const float* __restrict__ W,
                                       __bf16* __restrict__ wt) {
  int idx = blockIdx.x * blockDim.x + threadIdx.x;   // 0..16383
  int n = idx >> 7, k = idx & 127;
  wt[n * DIM + k] = (__bf16)W[k * DIM + n];
}

// ---------------- fused q/k/v projection GEMM (WMMA bf16 -> f32) ----------------
// One wave = one 16x16 output tile; block of 8 waves covers 16 rows x 128 cols.
// A: [M x 128] bf16 row-major activations. Wt: [128(n) x 128(k)] transposed bf16 weights.

__global__ __launch_bounds__(256) void gemm3_wmma_kernel(
    const __bf16* __restrict__ A,
    const __bf16* __restrict__ Wt0, const __bf16* __restrict__ Wt1, const __bf16* __restrict__ Wt2,
    const float* __restrict__ b0, const float* __restrict__ b1, const float* __restrict__ b2,
    float* __restrict__ O0, float* __restrict__ O1, float* __restrict__ O2) {
  const int lane  = threadIdx.x & 31;
  const int wave  = threadIdx.x >> 5;
  const int m0    = blockIdx.x * 16;
  const int n0    = wave * 16;
  const int l15   = lane & 15;
  const int khalf = lane >> 4;
  const int arow  = m0 + l15;   // A-frag: lane = matrix row
  const int bcol  = n0 + l15;   // B-frag: lane = matrix column

  v8f c0 = {}, c1 = {}, c2 = {};
  #pragma unroll
  for (int ks = 0; ks < 4; ++ks) {            // K = 128 in 4 steps of 32
    const int k0 = ks * 32;
    // A 16x32 bf16 layout (matches B): element e -> k = khalf*16+e
    v8bf alo = *(const v8bf*)(A + (size_t)arow * DIM + k0 + khalf * 16);
    v8bf ahi = *(const v8bf*)(A + (size_t)arow * DIM + k0 + khalf * 16 + 8);
    v16bf a;
    #pragma unroll
    for (int e = 0; e < 8; ++e) { a[e] = alo[e]; a[8 + e] = ahi[e]; }
    // B 32x16 bf16 layout: lane&15 = column, 16 contiguous k starting at k0 + khalf*16
    const int wofs = bcol * DIM + k0 + khalf * 16;
    v16bf f0 = *(const v16bf*)(Wt0 + wofs);
    v16bf f1 = *(const v16bf*)(Wt1 + wofs);
    v16bf f2 = *(const v16bf*)(Wt2 + wofs);
    c0 = __builtin_amdgcn_wmma_f32_16x16x32_bf16(false, a, false, f0, (short)0, c0, false, false);
    c1 = __builtin_amdgcn_wmma_f32_16x16x32_bf16(false, a, false, f1, (short)0, c1, false, false);
    c2 = __builtin_amdgcn_wmma_f32_16x16x32_bf16(false, a, false, f2, (short)0, c2, false, false);
  }
  // C/D layout: VGPR r -> M = m0 + khalf*8 + r, N = n0 + (lane&15)
  const int mbase = m0 + khalf * 8;
  const int ncol  = n0 + l15;
  const float bb0 = b0[ncol], bb1 = b1[ncol], bb2 = b2[ncol];
  #pragma unroll
  for (int r = 0; r < 8; ++r) {
    size_t o = (size_t)(mbase + r) * DIM + ncol;
    O0[o] = c0[r] + bb0;
    O1[o] = c1[r] + bb1;
    O2[o] = c2[r] + bb2;
  }
}

// ---------------- sparse (top-K) graph attention ----------------

__device__ __forceinline__ float wave_sum32(float x) {
  #pragma unroll
  for (int o = 16; o >= 1; o >>= 1) x += __shfl_xor(x, o, 32);
  return x;
}

template <int HEADS, bool RELU, bool LNRES>
__global__ __launch_bounds__(128) void attn_kernel(
    const float* __restrict__ q, const float* __restrict__ k, const float* __restrict__ v,
    const int* __restrict__ nbr, const float* __restrict__ xres,
    const float* __restrict__ gamma, const float* __restrict__ beta,
    float* __restrict__ out) {
  constexpr int HD = DIM / HEADS;             // 32 (4 heads) or 128 (1 head)
  __shared__ float sred[4];
  const int d    = threadIdx.x;               // output dim owned by this thread
  const int lane = d & 31;
  const int wv   = d >> 5;
  const int bi   = blockIdx.x;                // b*N + node
  const int node = bi & (NNODE - 1);
  const size_t base  = (size_t)bi * DIM;
  const size_t bbase = (size_t)(bi >> 11) * (size_t)NNODE * DIM;

  int nb[NBR];
  nb[0] = node;                               // self loop
  #pragma unroll
  for (int t = 0; t < TOPK; ++t) nb[1 + t] = nbr[node * TOPK + t];

  const float qd    = q[base + d];
  const float scale = rsqrtf((float)HD);
  float sc[NBR];
  #pragma unroll
  for (int j = 0; j < NBR; ++j) {
    float p = qd * k[bbase + (size_t)nb[j] * DIM + d];
    float s = wave_sum32(p);                  // head group == wave when HD==32
    if (HEADS == 1) {                         // cross-wave reduce for 128-dim head
      if (lane == 0) sred[wv] = s;
      __syncthreads();
      s = sred[0] + sred[1] + sred[2] + sred[3];
      __syncthreads();
    }
    sc[j] = s * scale;
  }
  // softmax over the 21 neighbors (masked entries are exactly zero weight)
  float m = sc[0];
  #pragma unroll
  for (int j = 1; j < NBR; ++j) m = fmaxf(m, sc[j]);
  float ssum = 0.f;
  #pragma unroll
  for (int j = 0; j < NBR; ++j) { sc[j] = __expf(sc[j] - m); ssum += sc[j]; }
  const float inv = 1.f / ssum;
  float o = 0.f;
  #pragma unroll
  for (int j = 0; j < NBR; ++j) o += (sc[j] * inv) * v[bbase + (size_t)nb[j] * DIM + d];
  if (RELU) o = fmaxf(o, 0.f);
  if (LNRES) {
    float y = o + xres[base + d];
    float s = wave_sum32(y);
    if (lane == 0) sred[wv] = s;
    __syncthreads();
    float mu = (sred[0] + sred[1] + sred[2] + sred[3]) * (1.f / DIM);
    __syncthreads();
    float df = y - mu;
    float s2 = wave_sum32(df * df);
    if (lane == 0) sred[wv] = s2;
    __syncthreads();
    float var = (sred[0] + sred[1] + sred[2] + sred[3]) * (1.f / DIM);
    out[base + d] = df * rsqrtf(var + LN_EPS) * gamma[d] + beta[d];
  } else {
    out[base + d] = o;
  }
}

// ---------------- driver ----------------

extern "C" void kernel_launch(void* const* d_in, const int* in_sizes, int n_in,
                              void* d_out, int out_size, void* d_ws, size_t ws_size,
                              hipStream_t stream) {
  (void)in_sizes; (void)n_in; (void)out_size; (void)ws_size;
  const float* x   = (const float*)d_in[0];
  const float* emb = (const float*)d_in[1];
  const float* W1[3] = { (const float*)d_in[2], (const float*)d_in[4], (const float*)d_in[6] };
  const float* B1[3] = { (const float*)d_in[3], (const float*)d_in[5], (const float*)d_in[7] };
  const float* W2[3] = { (const float*)d_in[8], (const float*)d_in[10], (const float*)d_in[12] };
  const float* B2[3] = { (const float*)d_in[9], (const float*)d_in[11], (const float*)d_in[13] };
  const float* g  = (const float*)d_in[14];
  const float* bt = (const float*)d_in[15];
  float* out = (float*)d_out;

  char* ws = (char*)d_ws;
  size_t off = 0;
  auto alloc = [&](size_t bytes) -> char* {
    char* p = ws + off;
    off += (bytes + 255) & ~(size_t)255;
    return p;
  };
  const size_t MTOT = (size_t)BATCH * NNODE;        // 16384 rows
  float*  nemb  = (float*)alloc((size_t)NNODE * 16 * 4);
  int*    nbr   = (int*)alloc((size_t)NNODE * TOPK * 4);
  __bf16* wt[6];
  for (int i = 0; i < 6; ++i) wt[i] = (__bf16*)alloc((size_t)DIM * DIM * 2);
  __bf16* actbf = (__bf16*)alloc(MTOT * DIM * 2);
  float*  qb    = (float*)alloc(MTOT * DIM * 4);
  float*  kb    = (float*)alloc(MTOT * DIM * 4);
  float*  vb    = (float*)alloc(MTOT * DIM * 4);
  float*  h1    = (float*)alloc(MTOT * DIM * 4);

  // 1. adjacency (batch-independent)
  normalize_emb_kernel<<<(NNODE + 255) / 256, 256, 0, stream>>>(emb, nemb);
  topk_kernel<<<NNODE, 256, 0, stream>>>(nemb, nbr);

  // 2. weight + activation precision prep
  for (int i = 0; i < 3; ++i) weight_to_bf16T_kernel<<<64, 256, 0, stream>>>(W1[i], wt[i]);
  for (int i = 0; i < 3; ++i) weight_to_bf16T_kernel<<<64, 256, 0, stream>>>(W2[i], wt[3 + i]);
  const int nact = (int)(MTOT * DIM);
  f32_to_bf16_kernel<<<nact / 256, 256, 0, stream>>>(x, actbf, nact);

  // 3. GAT layer 1: q/k/v projections (WMMA) + 4-head sparse attention + ReLU
  gemm3_wmma_kernel<<<(int)(MTOT / 16), 256, 0, stream>>>(
      actbf, wt[0], wt[1], wt[2], B1[0], B1[1], B1[2], qb, kb, vb);
  attn_kernel<4, true, false><<<(int)MTOT, 128, 0, stream>>>(
      qb, kb, vb, nbr, nullptr, nullptr, nullptr, h1);

  // 4. GAT layer 2: projections (WMMA) + 1-head sparse attention + residual + LayerNorm
  f32_to_bf16_kernel<<<nact / 256, 256, 0, stream>>>(h1, actbf, nact);
  gemm3_wmma_kernel<<<(int)(MTOT / 16), 256, 0, stream>>>(
      actbf, wt[3], wt[4], wt[5], B2[0], B2[1], B2[2], qb, kb, vb);
  attn_kernel<1, false, true><<<(int)MTOT, 128, 0, stream>>>(
      qb, kb, vb, nbr, x, g, bt, out);
}
